// SimpleMoELayer_47047071760974
// MI455X (gfx1250) — matrix operand order repaired and checked by reading, not compiled, not run on hardware
//
#include <hip/hip_runtime.h>
#include <hip/hip_bf16.h>

// ---------------- problem constants ----------------
constexpr int T_ = 2048;   // tokens
constexpr int D_ = 1024;   // d_model
constexpr int F_ = 4096;   // ffn dim
constexpr int E_ = 8;      // experts
constexpr int K_ = 2;      // top-k
constexpr int SLOTS = T_ * K_;   // 4096 routed (token,expert) pairs

// ---------------- WMMA types ----------------
typedef __bf16 bf16_t;
typedef __attribute__((ext_vector_type(4)))  __bf16 v4bf;
typedef __attribute__((ext_vector_type(8)))  __bf16 v8bf;
typedef __attribute__((ext_vector_type(16))) __bf16 v16bf;
typedef __attribute__((ext_vector_type(8)))  float  v8f;

__device__ __forceinline__ v16bf cat8(v8bf lo, v8bf hi) {
  return __builtin_shufflevector(lo, hi, 0,1,2,3,4,5,6,7,8,9,10,11,12,13,14,15);
}

// GEMM tiling: 256 threads = 8 waves. Block tile 128(M) x 128(N), K-chunk 32.
// Waves arranged 4(M) x 2(N); each wave computes a 32x64 strip:
// 2 A-fragments x 4 B-fragments = 8 WMMAs per K-step (12 b128 frag loads).
constexpr int TM  = 128;   // block M tile (slots)
constexpr int TNN = 128;   // block N tile
constexpr int TK  = 32;    // K chunk (bf16 WMMA K)
constexpr int LDA = 40;    // padded LDS row stride (bf16) -> conflict-free b128 frag loads
constexpr int LDB = 40;

// ---------------- kernel 0: zero small counters ----------------
__global__ void moe_zero_small(int* cnt_top1, int* route_counts, float* probsum_part) {
  int i = threadIdx.x;
  if (i < E_) { cnt_top1[i] = 0; route_counts[i] = 0; }
  if (i < E_ * E_) probsum_part[i] = 0.0f;  // 8 blocks x 8 experts
}

// ---------------- kernel 1: gating ----------------
__global__ void moe_gate(const float* __restrict__ x, const float* __restrict__ Wg,
                         const float* __restrict__ bg,
                         float* __restrict__ comb, int* __restrict__ top2,
                         int* __restrict__ cnt_top1, int* __restrict__ route_counts,
                         float* __restrict__ probsum_part) {
  __shared__ float sWg[D_ * E_];             // 32 KB
  __shared__ float red[256];
  const int tid = threadIdx.x;
  for (int i = tid; i < D_ * E_; i += 256) sWg[i] = Wg[i];
  __syncthreads();

  const int t = blockIdx.x * 256 + tid;      // exact: 8*256 == T_
  float l[E_];
  #pragma unroll
  for (int e = 0; e < E_; ++e) l[e] = bg[e];
  const float* xr = x + (size_t)t * D_;
  for (int d = 0; d < D_; ++d) {
    float xv = xr[d];
    #pragma unroll
    for (int e = 0; e < E_; ++e) l[e] = fmaf(xv, sWg[d * E_ + e], l[e]);
  }
  float m = l[0];
  #pragma unroll
  for (int e = 1; e < E_; ++e) m = fmaxf(m, l[e]);
  float p[E_]; float s = 0.0f;
  #pragma unroll
  for (int e = 0; e < E_; ++e) { p[e] = expf(l[e] - m); s += p[e]; }
  float inv = 1.0f / s;
  #pragma unroll
  for (int e = 0; e < E_; ++e) p[e] *= inv;
  // top-2 (first max wins on ties, matching jax.lax.top_k)
  int i1 = 0;
  #pragma unroll
  for (int e = 1; e < E_; ++e) if (p[e] > p[i1]) i1 = e;
  int i2 = (i1 == 0) ? 1 : 0;
  #pragma unroll
  for (int e = 0; e < E_; ++e) if (e != i1 && p[e] > p[i2]) i2 = e;

  #pragma unroll
  for (int e = 0; e < E_; ++e) comb[t * E_ + e] = 0.0f;
  comb[t * E_ + i1] = p[i1];
  comb[t * E_ + i2] = p[i2];
  top2[2 * t + 0] = i1;
  top2[2 * t + 1] = i2;
  atomicAdd(&cnt_top1[i1], 1);               // integer: deterministic
  atomicAdd(&route_counts[i1], 1);
  atomicAdd(&route_counts[i2], 1);

  // deterministic fixed-order tree reduction of prob sums (per-block partials)
  #pragma unroll
  for (int e = 0; e < E_; ++e) {
    red[tid] = p[e];
    __syncthreads();
    for (int s2 = 128; s2 > 0; s2 >>= 1) {
      if (tid < s2) red[tid] += red[tid + s2];
      __syncthreads();
    }
    if (tid == 0) probsum_part[blockIdx.x * E_ + e] = red[0];
    __syncthreads();
  }
}

// ---------------- kernel 2: offsets scan + aux loss ----------------
__global__ void moe_aux(const int* __restrict__ cnt_top1,
                        const float* __restrict__ probsum_part,
                        const int* __restrict__ route_counts,
                        int* __restrict__ offsets, float* __restrict__ aux_out) {
  if (threadIdx.x == 0 && blockIdx.x == 0) {
    float aux = 0.0f;
    int run = 0;
    for (int e = 0; e < E_; ++e) {
      offsets[e] = run;
      run += route_counts[e];
      float ps = 0.0f;
      for (int b = 0; b < E_; ++b) ps += probsum_part[b * E_ + e];   // fixed order
      aux += ((float)cnt_top1[e] / ((float)T_ + 1e-8f)) * (ps / (float)T_);
    }
    offsets[E_] = run;            // == SLOTS
    aux_out[0] = aux * (float)E_;
  }
}

// ---------------- kernel 3: deterministic per-expert compaction ----------------
__global__ void moe_scatter(const int* __restrict__ top2, const int* __restrict__ offsets,
                            int* __restrict__ token_list, int* __restrict__ slot_of) {
  const int e = blockIdx.x;
  __shared__ int flags[256];
  __shared__ int pref[257];
  int pos = offsets[e];
  for (int t0 = 0; t0 < T_; t0 += 256) {
    int t = t0 + threadIdx.x;
    int a = top2[2 * t], b = top2[2 * t + 1];
    int k = (a == e) ? 0 : ((b == e) ? 1 : -1);
    int f = (k >= 0) ? 1 : 0;
    flags[threadIdx.x] = f;
    __syncthreads();
    if (threadIdx.x == 0) {
      int run = 0;
      for (int i = 0; i < 256; ++i) { pref[i] = run; run += flags[i]; }
      pref[256] = run;
    }
    __syncthreads();
    if (f) {
      int slot = pos + pref[threadIdx.x];
      token_list[slot] = t;
      slot_of[2 * t + k] = slot;
    }
    pos += pref[256];
    __syncthreads();
  }
}

__device__ __forceinline__ v4bf cvt4(float a, float b, float c, float d) {
  v4bf r = { (bf16_t)a, (bf16_t)b, (bf16_t)c, (bf16_t)d };
  return r;
}

#define WMMA_BF16(A, B, C) \
  __builtin_amdgcn_wmma_f32_16x16x32_bf16(false, (A), false, (B), (short)0, (C), false, false)

// ---------------- kernel 4: FFN1  h = relu(x_gathered @ W1[e] + b1[e]) ----------------
__global__ void __launch_bounds__(256)
moe_ffn1(const float* __restrict__ x, const float* __restrict__ W1,
         const float* __restrict__ b1,
         const int* __restrict__ token_list, const int* __restrict__ offsets,
         bf16_t* __restrict__ h_ws) {
  const int e  = blockIdx.z;
  const int mt = blockIdx.y;
  const int nt = blockIdx.x;
  const int base = offsets[e];
  const int ne   = offsets[e + 1] - base;
  if (mt * TM >= ne) return;

  __shared__ __align__(16) bf16_t As[TM * LDA];    // 128 x 40
  __shared__ __align__(16) bf16_t Bt[TNN * LDB];   // 128 x 40 (transposed [n][k])
  __shared__ int toks[TM];

  const int tid = threadIdx.x;
  if (tid < TM) {
    int s = mt * TM + tid;
    toks[tid] = (s < ne) ? token_list[base + s] : 0;
  }
  __syncthreads();

  const float* W1e = W1 + (size_t)e * D_ * F_;
  const int f0 = nt * TNN;

  const int lane   = tid & 31;
  const int wave   = tid >> 5;
  const int m_off  = (wave >> 1) * 32;   // 4 M-strips of 32
  const int n_base = (wave & 1) * 64;    // 2 N-strips of 64
  const int lrow   = lane & 15;
  const int hi     = (lane >> 4) & 1;

  // staging coordinates: A row = tid/2, 16 consecutive k; B col = tid%128, 16 k
  const int ar = tid >> 1;
  const int ak = (tid & 1) * 16;
  const int bn = tid & 127;
  const int bk = (tid >> 7) * 16;

  const float* aptr = x + (size_t)toks[ar] * D_ + ak;

  float a_reg[16], b_reg[16];
  auto prefetch = [&](int k0) {
    const float4* px = (const float4*)(aptr + k0);
    #pragma unroll
    for (int q = 0; q < 4; ++q) {
      float4 v = px[q];
      a_reg[4 * q + 0] = v.x; a_reg[4 * q + 1] = v.y;
      a_reg[4 * q + 2] = v.z; a_reg[4 * q + 3] = v.w;
    }
    const float* wp = W1e + (size_t)(k0 + bk) * F_ + f0 + bn;
    #pragma unroll
    for (int jj = 0; jj < 16; ++jj) b_reg[jj] = wp[(size_t)jj * F_];
  };
  auto commit = [&]() {
    #pragma unroll
    for (int q = 0; q < 16; q += 4)
      *(v4bf*)(As + ar * LDA + ak + q) = cvt4(a_reg[q], a_reg[q+1], a_reg[q+2], a_reg[q+3]);
    #pragma unroll
    for (int q = 0; q < 16; q += 4)
      *(v4bf*)(Bt + bn * LDB + bk + q) = cvt4(b_reg[q], b_reg[q+1], b_reg[q+2], b_reg[q+3]);
  };

  v8f acc[2][4] = {};
  prefetch(0);
  for (int k0 = 0; k0 < D_; k0 += TK) {
    commit();                       // stores data prefetched for k0
    __syncthreads();
    if (k0 + TK < D_) prefetch(k0 + TK);   // overlap next loads with WMMA section
    const bf16_t* arow0 = As + (m_off + lrow) * LDA;
    const bf16_t* arow1 = arow0 + 16 * LDA;
    v16bf af0 = cat8(*(const v8bf*)(arow0 + hi * 8), *(const v8bf*)(arow0 + 16 + hi * 8));
    v16bf af1 = cat8(*(const v8bf*)(arow1 + hi * 8), *(const v8bf*)(arow1 + 16 + hi * 8));
    #pragma unroll
    for (int i = 0; i < 4; ++i) {
      const bf16_t* brow = Bt + (n_base + 16 * i + lrow) * LDB;
      v16bf bf = cat8(*(const v8bf*)(brow + hi * 8), *(const v8bf*)(brow + 16 + hi * 8));
      acc[0][i] = WMMA_BF16(af0, bf, acc[0][i]);
      acc[1][i] = WMMA_BF16(af1, bf, acc[1][i]);
    }
    __syncthreads();
  }

  // epilogue: C layout -> row M = j + 8*hi, col N = lane%16
  #pragma unroll
  for (int a = 0; a < 2; ++a) {
    const int rbase = mt * TM + m_off + a * 16 + hi * 8;
    #pragma unroll
    for (int i = 0; i < 4; ++i) {
      int fcol = f0 + n_base + 16 * i + lrow;
      float bias = b1[e * F_ + fcol];
      #pragma unroll
      for (int j = 0; j < 8; ++j) {
        int s = rbase + j;
        if (s < ne)
          h_ws[(size_t)(base + s) * F_ + fcol] = (bf16_t)fmaxf(acc[a][i][j] + bias, 0.0f);
      }
    }
  }
}

// ---------------- kernel 5: FFN2  y = h @ W2[e] + b2[e] ----------------
__global__ void __launch_bounds__(256)
moe_ffn2(const bf16_t* __restrict__ h_ws, const float* __restrict__ W2,
         const float* __restrict__ b2, const int* __restrict__ offsets,
         float* __restrict__ y_ws) {
  const int e  = blockIdx.z;
  const int mt = blockIdx.y;
  const int nt = blockIdx.x;
  const int base = offsets[e];
  const int ne   = offsets[e + 1] - base;
  if (mt * TM >= ne) return;

  __shared__ __align__(16) bf16_t As[TM * LDA];
  __shared__ __align__(16) bf16_t Bt[TNN * LDB];

  const int tid = threadIdx.x;
  const float* W2e = W2 + (size_t)e * F_ * D_;
  const int d0 = nt * TNN;

  const int lane   = tid & 31;
  const int wave   = tid >> 5;
  const int m_off  = (wave >> 1) * 32;
  const int n_base = (wave & 1) * 64;
  const int lrow   = lane & 15;
  const int hi     = (lane >> 4) & 1;

  const int ar = tid >> 1;
  const int ak = (tid & 1) * 16;
  const int bn = tid & 127;
  const int bk = (tid >> 7) * 16;

  int arow_g = base + mt * TM + ar;
  if (arow_g > SLOTS - 1) arow_g = SLOTS - 1;   // clamp (dead rows, stays in ws bounds)
  const bf16_t* aptr = h_ws + (size_t)arow_g * F_ + ak;

  v8bf a_regh[2];
  float b_reg[16];
  auto prefetch = [&](int k0) {
    a_regh[0] = *(const v8bf*)(aptr + k0);
    a_regh[1] = *(const v8bf*)(aptr + k0 + 8);
    const float* wp = W2e + (size_t)(k0 + bk) * D_ + d0 + bn;
    #pragma unroll
    for (int jj = 0; jj < 16; ++jj) b_reg[jj] = wp[(size_t)jj * D_];
  };
  auto commit = [&]() {
    *(v8bf*)(As + ar * LDA + ak)     = a_regh[0];
    *(v8bf*)(As + ar * LDA + ak + 8) = a_regh[1];
    #pragma unroll
    for (int q = 0; q < 16; q += 4)
      *(v4bf*)(Bt + bn * LDB + bk + q) = cvt4(b_reg[q], b_reg[q+1], b_reg[q+2], b_reg[q+3]);
  };

  v8f acc[2][4] = {};
  prefetch(0);
  for (int k0 = 0; k0 < F_; k0 += TK) {
    commit();
    __syncthreads();
    if (k0 + TK < F_) prefetch(k0 + TK);
    const bf16_t* arow0 = As + (m_off + lrow) * LDA;
    const bf16_t* arow1 = arow0 + 16 * LDA;
    v16bf af0 = cat8(*(const v8bf*)(arow0 + hi * 8), *(const v8bf*)(arow0 + 16 + hi * 8));
    v16bf af1 = cat8(*(const v8bf*)(arow1 + hi * 8), *(const v8bf*)(arow1 + 16 + hi * 8));
    #pragma unroll
    for (int i = 0; i < 4; ++i) {
      const bf16_t* brow = Bt + (n_base + 16 * i + lrow) * LDB;
      v16bf bf = cat8(*(const v8bf*)(brow + hi * 8), *(const v8bf*)(brow + 16 + hi * 8));
      acc[0][i] = WMMA_BF16(af0, bf, acc[0][i]);
      acc[1][i] = WMMA_BF16(af1, bf, acc[1][i]);
    }
    __syncthreads();
  }

  #pragma unroll
  for (int a = 0; a < 2; ++a) {
    const int rbase = mt * TM + m_off + a * 16 + hi * 8;
    #pragma unroll
    for (int i = 0; i < 4; ++i) {
      int dcol = d0 + n_base + 16 * i + lrow;
      float bias = b2[e * D_ + dcol];
      #pragma unroll
      for (int j = 0; j < 8; ++j) {
        int s = rbase + j;
        if (s < ne) y_ws[(size_t)(base + s) * D_ + dcol] = acc[a][i][j] + bias;
      }
    }
  }
}

// ---------------- kernel 6: combine (no atomics -> deterministic) ----------------
__global__ void moe_combine(const float* __restrict__ y_ws, const float* __restrict__ comb,
                            const int* __restrict__ top2, const int* __restrict__ slot_of,
                            float* __restrict__ out) {
  int idx = blockIdx.x * 256 + threadIdx.x;      // one float4 per thread
  if (idx >= T_ * D_ / 4) return;
  int t  = idx / (D_ / 4);
  int d4 = idx % (D_ / 4);
  float4 accv = make_float4(0.f, 0.f, 0.f, 0.f);
  #pragma unroll
  for (int k = 0; k < K_; ++k) {
    int e = top2[2 * t + k];
    float w = comb[t * E_ + e];
    int s = slot_of[2 * t + k];
    float4 y = ((const float4*)(y_ws + (size_t)s * D_))[d4];
    accv.x += w * y.x; accv.y += w * y.y; accv.z += w * y.z; accv.w += w * y.w;
  }
  ((float4*)out)[idx] = accv;
}

// ---------------- host launcher ----------------
extern "C" void kernel_launch(void* const* d_in, const int* in_sizes, int n_in,
                              void* d_out, int out_size, void* d_ws, size_t ws_size,
                              hipStream_t stream) {
  const float* x  = (const float*)d_in[0];
  const float* Wg = (const float*)d_in[1];
  const float* bg = (const float*)d_in[2];
  const float* W1 = (const float*)d_in[3];
  const float* b1 = (const float*)d_in[4];
  const float* W2 = (const float*)d_in[5];
  const float* b2 = (const float*)d_in[6];
  float* out = (float*)d_out;

  // workspace layout (256B aligned)
  char* ws = (char*)d_ws;
  size_t o = 0;
  auto take = [&](size_t bytes) -> void* {
    o = (o + 255) & ~(size_t)255;
    void* p = ws + o;
    o += bytes;
    return p;
  };
  float* comb         = (float*)take(sizeof(float) * T_ * E_);
  int*   top2         = (int*)  take(sizeof(int) * 2 * T_);
  int*   slot_of      = (int*)  take(sizeof(int) * 2 * T_);
  int*   token_list   = (int*)  take(sizeof(int) * SLOTS);
  int*   cnt_top1     = (int*)  take(sizeof(int) * E_);
  int*   route_counts = (int*)  take(sizeof(int) * E_);
  float* probsum_part = (float*)take(sizeof(float) * E_ * E_);
  int*   offsets      = (int*)  take(sizeof(int) * (E_ + 1));
  bf16_t* h_ws        = (bf16_t*)take(sizeof(bf16_t) * (size_t)SLOTS * F_);
  float*  y_ws        = (float*) take(sizeof(float)  * (size_t)SLOTS * D_);
  (void)ws_size; (void)in_sizes; (void)n_in; (void)out_size;

  moe_zero_small<<<1, 128, 0, stream>>>(cnt_top1, route_counts, probsum_part);
  moe_gate<<<T_ / 256, 256, 0, stream>>>(x, Wg, bg, comb, top2, cnt_top1,
                                         route_counts, probsum_part);
  moe_aux<<<1, 32, 0, stream>>>(cnt_top1, probsum_part, route_counts, offsets,
                                out + (size_t)T_ * D_);
  moe_scatter<<<E_, 256, 0, stream>>>(top2, offsets, token_list, slot_of);
  // worst case an expert owns all T_ tokens -> T_/TM M-tiles (extra blocks early-exit)
  moe_ffn1<<<dim3(F_ / TNN, T_ / TM, E_), 256, 0, stream>>>(x, W1, b1, token_list,
                                                            offsets, h_ws);
  moe_ffn2<<<dim3(D_ / TNN, T_ / TM, E_), 256, 0, stream>>>(h_ws, W2, b2, offsets, y_ws);
  moe_combine<<<(T_ * D_ / 4 + 255) / 256, 256, 0, stream>>>(y_ws, comb, top2, slot_of, out);
}
